// GLNFeat_11905649344755
// MI455X (gfx1250) — matrix operand order repaired and checked by reading, not compiled, not run on hardware
//
#include <hip/hip_runtime.h>

typedef __attribute__((ext_vector_type(16))) _Float16 v16h;
typedef __attribute__((ext_vector_type(8)))  float    v8f;

#define BATCH 4096
#define EMBD 64
#define ITEM_COLS 10242

__device__ __forceinline__ float lrelu(float v) { return v >= 0.f ? v : 0.01f * v; }

// ---------------------------------------------------------------------------
// Fold kernel (f32): out[r*64+d] = sum_k T[r*64+k] * W[k*64+d]
// Folds embedding tables through user_W/item_W 64x64 blocks.
// ---------------------------------------------------------------------------
__global__ void fold_f32_kernel(const float* __restrict__ T, int rows,
                                const float* __restrict__ W,
                                float* __restrict__ out) {
  int tid = blockIdx.x * blockDim.x + threadIdx.x;
  if (tid >= rows * EMBD) return;
  int r = tid >> 6, d = tid & 63;
  const float* trow = T + r * EMBD;
  float s = 0.f;
#pragma unroll 8
  for (int k = 0; k < EMBD; ++k) s = fmaf(trow[k], W[k * EMBD + d], s);
  out[tid] = s;
}

// ---------------------------------------------------------------------------
// Fold + swizzle to f16 WMMA B-fragment order.
// Folded weight Wf[K,n] = sum_c FW[K,c] * W[c*64+n], K in [0, ktiles*32),
// zero-padded past 'width'. B-frag layout (16x16x32 f16, 32x16 KxN tile):
//   lane L holds col N = L%16 ; half j holds K = (j<8 ? j : j+8) + (L<16 ? 0 : 8)
// Stored so each lane's 16 halves are contiguous (32B per lane):
//   out[ ((kt*4 + nt)*32 + lane)*16 + j ]
// ---------------------------------------------------------------------------
__global__ void fold_swz_f16_kernel(const float* __restrict__ FW, int width, int ktiles,
                                    const float* __restrict__ W,
                                    _Float16* __restrict__ out) {
  int tid = blockIdx.x * blockDim.x + threadIdx.x;
  if (tid >= ktiles * 2048) return;
  int j    =  tid        & 15;
  int lane = (tid >> 4)  & 31;
  int nt   = (tid >> 9)  & 3;
  int kt   =  tid >> 11;
  int K = kt * 32 + (j < 8 ? j : j + 8) + ((lane & 16) ? 8 : 0);
  int n = nt * 16 + (lane & 15);
  float s = 0.f;
  if (K < width) {
    const float* frow = FW + K * EMBD;
#pragma unroll 8
    for (int c = 0; c < EMBD; ++c) s = fmaf(frow[c], W[c * EMBD + n], s);
  }
  out[((kt * 4 + nt) * 32 + lane) * 16 + j] = (_Float16)s;
}

// ---------------------------------------------------------------------------
// User branch (pure lookups after folding) + edge embedding.
// ---------------------------------------------------------------------------
__global__ void user_edge_kernel(const int* __restrict__ ue, const int* __restrict__ ee,
                                 const float* __restrict__ gF, const float* __restrict__ aF,
                                 const float* __restrict__ oF, const float* __restrict__ arF,
                                 const float* __restrict__ user_b,
                                 const float* __restrict__ edges_tab,
                                 float* __restrict__ x, float* __restrict__ edge) {
  int tid = blockIdx.x * blockDim.x + threadIdx.x;
  if (tid >= BATCH * EMBD) return;
  int b = tid >> 6, d = tid & 63;
  const int* u = ue + b * 4;
  float xv = gF[u[0] * EMBD + d] + aF[u[1] * EMBD + d] +
             oF[u[2] * EMBD + d] + arF[u[3] * EMBD + d] + user_b[d];
  x[tid] = xv * 0.12f;
  edge[tid] = edges_tab[ee[b] * EMBD + d] * 0.12f;
}

// ---------------------------------------------------------------------------
// One wave's K-strided share of a multi-hot field GEMM.
// A-frag layout: lane L = row L%16; half j -> K = (j<8?j:j+8) + (L<16?0:8)
// Produces partial 16x64 accumulators (f0..f3) and partial count.
// ---------------------------------------------------------------------------
__device__ __forceinline__ void field_mm_part(const int* __restrict__ arow,
                                              int base, int width, int nkt, int kt0,
                                              const _Float16* __restrict__ wbuf,
                                              int lane, int hs,
                                              v8f& f0, v8f& f1, v8f& f2, v8f& f3,
                                              float& cnt) {
  for (int kt = kt0; kt < nkt; kt += 8) {
    if (kt + 8 < nkt)  // prefetch next strided tile's row segment (128B line)
      __builtin_prefetch((const void*)(arow + base + (kt + 8) * 32), 0, 3);
    int k0 = kt * 32;
    int kb = k0 + hs * 8;      // field-relative start col for this lane half
    v16h af;
    int csum;
    if (k0 + 32 <= width) {    // wave-uniform fast path: 8B-aligned int2 loads
      const int* p = arow + base + kb;
      int2 q0 = *(const int2*)(p);
      int2 q1 = *(const int2*)(p + 2);
      int2 q2 = *(const int2*)(p + 4);
      int2 q3 = *(const int2*)(p + 6);
      int2 q4 = *(const int2*)(p + 16);
      int2 q5 = *(const int2*)(p + 18);
      int2 q6 = *(const int2*)(p + 20);
      int2 q7 = *(const int2*)(p + 22);
      af[0]  = (_Float16)(float)q0.x; af[1]  = (_Float16)(float)q0.y;
      af[2]  = (_Float16)(float)q1.x; af[3]  = (_Float16)(float)q1.y;
      af[4]  = (_Float16)(float)q2.x; af[5]  = (_Float16)(float)q2.y;
      af[6]  = (_Float16)(float)q3.x; af[7]  = (_Float16)(float)q3.y;
      af[8]  = (_Float16)(float)q4.x; af[9]  = (_Float16)(float)q4.y;
      af[10] = (_Float16)(float)q5.x; af[11] = (_Float16)(float)q5.y;
      af[12] = (_Float16)(float)q6.x; af[13] = (_Float16)(float)q6.y;
      af[14] = (_Float16)(float)q7.x; af[15] = (_Float16)(float)q7.y;
      csum = q0.x + q0.y + q1.x + q1.y + q2.x + q2.y + q3.x + q3.y +
             q4.x + q4.y + q5.x + q5.y + q6.x + q6.y + q7.x + q7.y;
    } else {                   // boundary tile: per-element predicated loads
      csum = 0;
#pragma unroll
      for (int jj = 0; jj < 8; ++jj) {
        int kkA = kb + jj;
        int kkB = kb + 16 + jj;
        int v0 = (kkA < width) ? arow[base + kkA] : 0;
        int v1 = (kkB < width) ? arow[base + kkB] : 0;
        af[jj]     = (_Float16)(float)v0;
        af[jj + 8] = (_Float16)(float)v1;
        csum += v0 + v1;
      }
    }
    cnt += (float)csum;
    const _Float16* wb = wbuf + (kt * 128 + lane) * 16;  // kt*4*32 lanes * 16 halves
    v16h b0 = *(const v16h*)(wb);
    v16h b1 = *(const v16h*)(wb + 512);
    v16h b2 = *(const v16h*)(wb + 1024);
    v16h b3 = *(const v16h*)(wb + 1536);
    f0 = __builtin_amdgcn_wmma_f32_16x16x32_f16(false, af, false, b0, (short)0, f0, false, false);
    f1 = __builtin_amdgcn_wmma_f32_16x16x32_f16(false, af, false, b1, (short)0, f1, false, false);
    f2 = __builtin_amdgcn_wmma_f32_16x16x32_f16(false, af, false, b2, (short)0, f2, false, false);
    f3 = __builtin_amdgcn_wmma_f32_16x16x32_f16(false, af, false, b3, (short)0, f3, false, false);
  }
}

// ---------------------------------------------------------------------------
// Item branch: y = (rate'[idx] + sum_f (1/cnt_f) A01_f @ Wf' + item_b) * 0.12
// One block (8 waves) per 16-row tile; waves K-split each field 8-ways,
// partial tiles + counts reduced in LDS per field (normalization is linear,
// so scaling by 1/cnt after the cross-wave reduction is exact).
// ---------------------------------------------------------------------------
__global__ void __launch_bounds__(256)
item_y_kernel(const int* __restrict__ item,
              const _Float16* __restrict__ wg, const _Float16* __restrict__ wd,
              const _Float16* __restrict__ wa,
              const float* __restrict__ rateF, const float* __restrict__ item_b,
              float* __restrict__ y) {
  __shared__ float part[8][1024];  // per-wave partial 16x64 tiles
  __shared__ float cnts[8][16];    // per-wave partial row counts
  __shared__ float cinv[16];       // 1 / total row count

  const int t    = threadIdx.x;
  const int lane = t & 31;
  const int wave = t >> 5;
  const int r0   = blockIdx.x * 16;
  const int hs   = (lane >> 4) & 1;
  const int* arow = item + (size_t)(r0 + (lane & 15)) * ITEM_COLS;

  float facc[4] = {0.f, 0.f, 0.f, 0.f};  // this thread's 4 reduced elements

  const int fbase[3] = {1, 26, 2212};
  const int fwid[3]  = {25, 2186, 8030};
  const int fkt[3]   = {1, 69, 251};
  const _Float16* fw[3] = {wg, wd, wa};

#pragma unroll
  for (int f = 0; f < 3; ++f) {
    v8f f0 = {}, f1 = {}, f2 = {}, f3 = {};
    float cnt = 0.f;
    field_mm_part(arow, fbase[f], fwid[f], fkt[f], wave, fw[f], lane, hs,
                  f0, f1, f2, f3, cnt);

    // partial count: lanes L and L+16 cover complementary K; row = L%16
    cnt += __shfl_xor(cnt, 16, 32);
    if (lane < 16) cnts[wave][lane] = cnt;

    // store partial tile: element e = (nt*8 + i)*32 + lane
#pragma unroll
    for (int i = 0; i < 8; ++i) {
      part[wave][(0 * 8 + i) * 32 + lane] = f0[i];
      part[wave][(1 * 8 + i) * 32 + lane] = f1[i];
      part[wave][(2 * 8 + i) * 32 + lane] = f2[i];
      part[wave][(3 * 8 + i) * 32 + lane] = f3[i];
    }
    __syncthreads();

    if (t < 16) {
      float c = 0.f;
#pragma unroll
      for (int w = 0; w < 8; ++w) c += cnts[w][t];
      cinv[t] = 1.0f / c;   // matches reference (inf on empty row)
    }
    __syncthreads();

    // cross-wave reduce + scale; each thread owns 4 elements e = t + 256q
#pragma unroll
    for (int q = 0; q < 4; ++q) {
      int e  = t + 256 * q;
      int el = e & 31;
      int m  = ((e >> 5) & 7) + ((el & 16) ? 8 : 0);  // row within tile
      float s = 0.f;
#pragma unroll
      for (int w = 0; w < 8; ++w) s += part[w][e];
      facc[q] += s * cinv[m];
    }
    __syncthreads();  // before next field overwrites 'part'
  }

  // epilogue: e = t + 256q -> (nt = q, i, el); row/col; add rate'+bias, *0.12
#pragma unroll
  for (int q = 0; q < 4; ++q) {
    int e   = t + 256 * q;
    int el  = e & 31;
    int i   = (e >> 5) & 7;
    int nt  = e >> 8;
    int m   = i + ((el & 16) ? 8 : 0);
    int rr  = r0 + m;
    int col = nt * 16 + (el & 15);
    int rid = item[(size_t)rr * ITEM_COLS];  // rate index (col 0)
    y[rr * EMBD + col] = (facc[q] + rateF[rid * EMBD + col] + item_b[col]) * 0.12f;
  }
}

// ---------------------------------------------------------------------------
// Per-row dot products: sc = [x_xx, x_xy_x, x_xy_y, y_yy, y_yx_y, y_yx_x]
// ---------------------------------------------------------------------------
__global__ void scalars_kernel(const float* __restrict__ x, const float* __restrict__ y,
                               const float* __restrict__ edge,
                               const float* __restrict__ uu_w, const float* __restrict__ ui_w,
                               const float* __restrict__ iu_w, const float* __restrict__ ii_w,
                               float* __restrict__ sc) {
  int b = blockIdx.x * blockDim.x + threadIdx.x;
  if (b >= BATCH) return;
  float xx = 0, xyx = 0, xyy = 0, yy = 0, yxy = 0, yxx = 0;
#pragma unroll 8
  for (int d = 0; d < EMBD; ++d) {
    float e  = edge[b * EMBD + d];
    float xe = x[b * EMBD + d] * e;
    float ye = y[b * EMBD + d] * e;
    xx  = fmaf(xe, uu_w[d], xx);
    xyx = fmaf(xe, ui_w[d], xyx);
    xyy = fmaf(ye, ui_w[d], xyy);
    yy  = fmaf(ye, ii_w[d], yy);
    yxy = fmaf(ye, iu_w[d], yxy);
    yxx = fmaf(xe, iu_w[d], yxx);
  }
  sc[b] = xx;             sc[BATCH + b] = xyx;     sc[2 * BATCH + b] = xyy;
  sc[3 * BATCH + b] = yy; sc[4 * BATCH + b] = yxy; sc[5 * BATCH + b] = yxx;
}

// ---------------------------------------------------------------------------
// Batch reductions: r1 = x_xx@x, r2 = x_xy_y@y, r3 = y_yy@y, r4 = y_yx_x@x
// ---------------------------------------------------------------------------
__global__ void colreduce_kernel(const float* __restrict__ x, const float* __restrict__ y,
                                 const float* __restrict__ sc, float* __restrict__ r) {
  int d = blockIdx.x, t = threadIdx.x;
  const float* xx  = sc;
  const float* xyy = sc + 2 * BATCH;
  const float* yy  = sc + 3 * BATCH;
  const float* yxx = sc + 5 * BATCH;
  float p0 = 0, p1 = 0, p2 = 0, p3 = 0;
  for (int b = t; b < BATCH; b += 256) {
    float xv = x[b * EMBD + d], yv = y[b * EMBD + d];
    p0 = fmaf(xx[b],  xv, p0);
    p1 = fmaf(xyy[b], yv, p1);
    p2 = fmaf(yy[b],  yv, p2);
    p3 = fmaf(yxx[b], xv, p3);
  }
  __shared__ float red[4][256];
  red[0][t] = p0; red[1][t] = p1; red[2][t] = p2; red[3][t] = p3;
  __syncthreads();
  for (int s = 128; s > 0; s >>= 1) {
    if (t < s)
      for (int q = 0; q < 4; ++q) red[q][t] += red[q][t + s];
    __syncthreads();
  }
  if (t == 0) {
    r[d] = red[0][0]; r[64 + d] = red[1][0];
    r[128 + d] = red[2][0]; r[192 + d] = red[3][0];
  }
}

// s_v = r_v @ W1_v  (four 64-vec x 64x64 mats)
__global__ void smallmat_kernel(const float* __restrict__ r,
                                const float* __restrict__ uu_W1, const float* __restrict__ ui_W1,
                                const float* __restrict__ ii_W1, const float* __restrict__ iu_W1,
                                float* __restrict__ s) {
  int t = threadIdx.x;
  int v = t >> 6, n = t & 63;
  const float* W = (v == 0) ? uu_W1 : (v == 1) ? ui_W1 : (v == 2) ? ii_W1 : iu_W1;
  const float* rv = r + v * 64;
  float acc = 0.f;
#pragma unroll 8
  for (int k = 0; k < EMBD; ++k) acc = fmaf(rv[k], W[k * EMBD + n], acc);
  s[t] = acc;
}

// ---------------------------------------------------------------------------
// Final combine. out = [x_out | y_out | edge_out] flat.
// ---------------------------------------------------------------------------
__global__ void final_kernel(const float* __restrict__ x, const float* __restrict__ y,
                             const float* __restrict__ edge, const float* __restrict__ sc,
                             const float* __restrict__ s, const float* __restrict__ edge_W,
                             float* __restrict__ out) {
  int tid = blockIdx.x * blockDim.x + threadIdx.x;
  if (tid >= BATCH * EMBD) return;
  int b = tid >> 6, d = tid & 63;
  const float* erow = edge + b * EMBD;
  float em = 0.f;
#pragma unroll 8
  for (int k = 0; k < EMBD; ++k) em = fmaf(erow[k], edge_W[k * EMBD + d], em);
  float xx  = sc[b];
  float xyx = sc[BATCH + b];
  float yy  = sc[3 * BATCH + b];
  float yxy = sc[4 * BATCH + b];
  float xbuy = (lrelu(xx * s[d]) + lrelu(xyx * s[64 + d])) * 0.5f;
  float ybuy = (lrelu(yy * s[128 + d]) + lrelu(yxy * s[192 + d])) * 0.5f;
  out[tid]                    = (xbuy + x[tid]) * 0.5f;
  out[BATCH * EMBD + tid]     = (ybuy + y[tid]) * 0.5f;
  out[2 * BATCH * EMBD + tid] = (lrelu(em) + erow[d]) * 0.5f;
}

static inline int cdiv(int a, int b) { return (a + b - 1) / b; }

extern "C" void kernel_launch(void* const* d_in, const int* in_sizes, int n_in,
                              void* d_out, int out_size, void* d_ws, size_t ws_size,
                              hipStream_t stream) {
  const int*   user_emb   = (const int*)d_in[0];
  const int*   item_emb   = (const int*)d_in[1];
  const int*   edge_emb   = (const int*)d_in[2];
  const float* gender_tab = (const float*)d_in[3];
  const float* age_tab    = (const float*)d_in[4];
  const float* occ_tab    = (const float*)d_in[5];
  const float* area_tab   = (const float*)d_in[6];
  const float* user_W     = (const float*)d_in[7];
  const float* user_b     = (const float*)d_in[8];
  const float* rate_tab   = (const float*)d_in[9];
  const float* genre_W    = (const float*)d_in[10];
  const float* director_W = (const float*)d_in[11];
  const float* actor_W    = (const float*)d_in[12];
  const float* item_W     = (const float*)d_in[13];
  const float* item_b     = (const float*)d_in[14];
  const float* edges_tab  = (const float*)d_in[15];
  const float* uu_w       = (const float*)d_in[16];
  const float* ui_w       = (const float*)d_in[17];
  const float* iu_w       = (const float*)d_in[18];
  const float* ii_w       = (const float*)d_in[19];
  const float* edge_W     = (const float*)d_in[20];
  const float* uu_W1      = (const float*)d_in[21];
  const float* ui_W1      = (const float*)d_in[22];
  const float* iu_W1      = (const float*)d_in[23];
  const float* ii_W1      = (const float*)d_in[24];

  size_t off = 0;
  auto wsalloc = [&](size_t bytes) -> char* {
    char* p = (char*)d_ws + off;
    off += (bytes + 255) & ~(size_t)255;
    return p;
  };
  float*    genderF = (float*)wsalloc(98 * 64 * 4);
  float*    ageF    = (float*)wsalloc(7 * 64 * 4);
  float*    occF    = (float*)wsalloc(21 * 64 * 4);
  float*    areaF   = (float*)wsalloc(3402 * 64 * 4);
  float*    rateF   = (float*)wsalloc(6 * 64 * 4);
  _Float16* wg      = (_Float16*)wsalloc(1 * 2048 * 2);
  _Float16* wd      = (_Float16*)wsalloc(69 * 2048 * 2);
  _Float16* wa      = (_Float16*)wsalloc(251 * 2048 * 2);
  float*    xbuf    = (float*)wsalloc(BATCH * EMBD * 4);
  float*    ybuf    = (float*)wsalloc(BATCH * EMBD * 4);
  float*    ebuf    = (float*)wsalloc(BATCH * EMBD * 4);
  float*    scbuf   = (float*)wsalloc(6 * BATCH * 4);
  float*    rvec    = (float*)wsalloc(4 * 64 * 4);
  float*    svec    = (float*)wsalloc(4 * 64 * 4);
  (void)in_sizes; (void)n_in; (void)out_size; (void)ws_size;

  // Fold embedding tables through user_W / item_W 64x64 blocks.
  fold_f32_kernel<<<cdiv(98 * 64, 256), 256, 0, stream>>>(gender_tab, 98, user_W + 0,     genderF);
  fold_f32_kernel<<<cdiv(7 * 64, 256),  256, 0, stream>>>(age_tab,     7, user_W + 4096,  ageF);
  fold_f32_kernel<<<cdiv(21 * 64, 256), 256, 0, stream>>>(occ_tab,    21, user_W + 8192,  occF);
  fold_f32_kernel<<<cdiv(3402 * 64, 256), 256, 0, stream>>>(area_tab, 3402, user_W + 12288, areaF);
  fold_f32_kernel<<<cdiv(6 * 64, 256),  256, 0, stream>>>(rate_tab,    6, item_W + 0,     rateF);

  // Fold + swizzle multi-hot field weights to f16 B-fragment order.
  fold_swz_f16_kernel<<<cdiv(1 * 2048, 256),   256, 0, stream>>>(genre_W,      25,   1, item_W + 4096,  wg);
  fold_swz_f16_kernel<<<cdiv(69 * 2048, 256),  256, 0, stream>>>(director_W, 2186,  69, item_W + 8192,  wd);
  fold_swz_f16_kernel<<<cdiv(251 * 2048, 256), 256, 0, stream>>>(actor_W,    8030, 251, item_W + 12288, wa);

  user_edge_kernel<<<cdiv(BATCH * EMBD, 256), 256, 0, stream>>>(
      user_emb, edge_emb, genderF, ageF, occF, areaF, user_b, edges_tab, xbuf, ebuf);

  // 256 blocks x 8 waves = 2048 waves; HBM-stream of item_emb w/ WMMA.
  item_y_kernel<<<256, 256, 0, stream>>>(item_emb, wg, wd, wa, rateF, item_b, ybuf);

  scalars_kernel<<<cdiv(BATCH, 256), 256, 0, stream>>>(
      xbuf, ybuf, ebuf, uu_w, ui_w, iu_w, ii_w, scbuf);

  colreduce_kernel<<<64, 256, 0, stream>>>(xbuf, ybuf, scbuf, rvec);

  smallmat_kernel<<<1, 256, 0, stream>>>(rvec, uu_W1, ui_W1, ii_W1, iu_W1, svec);

  final_kernel<<<cdiv(BATCH * EMBD, 256), 256, 0, stream>>>(
      xbuf, ybuf, ebuf, scbuf, svec, edge_W, (float*)d_out);
}
